// DenseNet_1228360646808
// MI455X (gfx1250) — matrix operand order repaired and checked by reading, not compiled, not run on hardware
//
#include <hip/hip_runtime.h>
#include <hip/hip_bf16.h>

// ---------------------------------------------------------------------------
// Types for CDNA5 WMMA (gfx1250, wave32)
// ---------------------------------------------------------------------------
typedef __attribute__((ext_vector_type(16))) __bf16 v16bf;
typedef __attribute__((ext_vector_type(8)))  __bf16 v8bf;   // 16 bytes -> global_load_b128
typedef __attribute__((ext_vector_type(8)))  float  v8f;

union AFrag { v16bf v; v8bf h[2]; };

// ---------------------------------------------------------------------------
// Weight convert + pad: W f32 [O,I] -> Wb bf16 [Np, Kp]
//   rows o >= O are zero, cols k >= I are zero  (Np = pad64(O), Kp = pad32(I))
// so the GEMM needs NO guards on B-fragment loads at all.
// ---------------------------------------------------------------------------
__global__ void convw_kernel(const float* __restrict__ W, __bf16* __restrict__ Wb,
                             int O, int I, int Np, int Kp) {
    int idx = blockIdx.x * 256 + threadIdx.x;
    int total = Np * Kp;
    if (idx >= total) return;
    int o = idx / Kp;
    int k = idx - o * Kp;
    Wb[idx] = (o < O && k < I) ? (__bf16)W[(size_t)o * I + k] : (__bf16)0.0f;
}

// ---------------------------------------------------------------------------
// Embedding gather: tokens [T*B] -> X bf16 [8192, ldx], features 0..511
// ---------------------------------------------------------------------------
__global__ void embed_kernel(const int* __restrict__ tok, const float* __restrict__ E,
                             __bf16* __restrict__ X, int ldx) {
    int r = blockIdx.x;                       // 0..8191  (row = t*128 + b)
    int v = tok[r];
    const float* e = E + (size_t)v * 512;
    __bf16* x = X + (size_t)r * ldx;
    for (int c = threadIdx.x; c < 512; c += 256) x[c] = (__bf16)e[c];
}

// ---------------------------------------------------------------------------
// WMMA bf16 GEMM:  C[M,N] = A[M,K] * Bw[N,K]^T + bias   (C row-major f32)
//   A : bf16, row stride lda (uses cols [0,K))
//   Bw: bf16, row-major [Np, ldb]; Np mult-of-64 zero-padded rows,
//       ldb = pad32(K) zero-padded cols  -> B loads are unconditional.
// One wave computes a 64x64 tile: 4 A-frags + 4 B-frags -> 16 v_wmma / K-chunk
// (1.0 b128 load per WMMA; halves L2 re-fetch vs 32-wide M tiles).
// Fragment layouts per CDNA5 ISA 7.12.2 (wave32).
// ---------------------------------------------------------------------------
__global__ void __launch_bounds__(128)
gemm_wmma_kernel(const __bf16* __restrict__ A, int lda,
                 const __bf16* __restrict__ Bw, int ldb,
                 const float* __restrict__ bias,
                 float* __restrict__ C, int ldc,
                 int M, int N, int K) {
    const int wave   = (blockIdx.x * blockDim.x + threadIdx.x) >> 5;
    const int lane   = threadIdx.x & 31;
    const int tilesN = (N + 63) >> 6;
    const int tilesM = M >> 6;
    const int tm = wave / tilesN;
    const int tn = wave - tm * tilesN;
    if (tm >= tilesM) return;

    const int row0 = tm * 64, col0 = tn * 64;
    const int ml = lane & 15;           // A: row-in-tile / B: col-in-tile / D: col
    const int hi = lane >> 4;

    const __bf16* Arow[4];
#pragma unroll
    for (int mi = 0; mi < 4; ++mi)
        Arow[mi] = A + (size_t)(row0 + 16 * mi + ml) * lda;
    const __bf16* Brow[4];
#pragma unroll
    for (int q = 0; q < 4; ++q)
        Brow[q] = Bw + (size_t)(col0 + 16 * q + ml) * ldb;   // always in padded range

    v8f acc[4][4];
#pragma unroll
    for (int mi = 0; mi < 4; ++mi)
#pragma unroll
        for (int q = 0; q < 4; ++q) acc[mi][q] = v8f{};

    const int kfull = K & ~31;
    for (int k = 0; k < kfull; k += 32) {
        AFrag a[4], b[4];
#pragma unroll
        for (int mi = 0; mi < 4; ++mi) {
            // A: K runs [k+8*hi, +8) and [k+16+8*hi, +8)  -> two b128 loads
            a[mi].h[0] = *(const v8bf*)(Arow[mi] + k + 8 * hi);
            a[mi].h[1] = *(const v8bf*)(Arow[mi] + k + 16 + 8 * hi);
        }
#pragma unroll
        for (int q = 0; q < 4; ++q) {
            // B: K runs [k+8*hi, +8) and [k+16+8*hi, +8) -> two b128 loads (fully padded weights)
            b[q].h[0] = *(const v8bf*)(Brow[q] + k + 8 * hi);
            b[q].h[1] = *(const v8bf*)(Brow[q] + k + 16 + 8 * hi);
        }
#pragma unroll
        for (int mi = 0; mi < 4; ++mi)
#pragma unroll
            for (int q = 0; q < 4; ++q)
                acc[mi][q] = __builtin_amdgcn_wmma_f32_16x16x32_bf16(
                    false, a[mi].v, false, b[q].v, (short)0, acc[mi][q], false, false);
    }

    if (kfull < K) {   // K tail (K % 32 != 0): guarded element loads for A only
        AFrag a[4], b[4];
#pragma unroll
        for (int e = 0; e < 16; ++e) {
            int ka = kfull + 8 * hi + ((e < 8) ? e : (8 + e));   // 8+e == 16+(e-8)
            __bf16 z = (__bf16)0.0f;
#pragma unroll
            for (int mi = 0; mi < 4; ++mi)
                a[mi].v[e] = (ka < K) ? Arow[mi][ka] : z;
        }
#pragma unroll
        for (int q = 0; q < 4; ++q) {
            b[q].h[0] = *(const v8bf*)(Brow[q] + kfull + 8 * hi);
            b[q].h[1] = *(const v8bf*)(Brow[q] + kfull + 16 + 8 * hi);
        }
#pragma unroll
        for (int mi = 0; mi < 4; ++mi)
#pragma unroll
            for (int q = 0; q < 4; ++q)
                acc[mi][q] = __builtin_amdgcn_wmma_f32_16x16x32_bf16(
                    false, a[mi].v, false, b[q].v, (short)0, acc[mi][q], false, false);
    }

    // D layout: lane col = ml (+16*q), row = r + 8*hi (+16*mi)
#pragma unroll
    for (int q = 0; q < 4; ++q) {
        const int col = col0 + 16 * q + ml;
        if (col < N) {
            const float bv = bias[col];
#pragma unroll
            for (int mi = 0; mi < 4; ++mi) {
#pragma unroll
                for (int r = 0; r < 8; ++r) {
                    int rr = row0 + 16 * mi + r + 8 * hi;
                    C[(size_t)rr * ldc + col] = acc[mi][q][r] + bv;
                }
            }
        }
    }
}

// ---------------------------------------------------------------------------
// Block reduction over 128 threads (4 wave32)
// ---------------------------------------------------------------------------
static __device__ __forceinline__ float blockSum128(float v, float* s) {
    v += __shfl_xor(v, 16, 32);
    v += __shfl_xor(v,  8, 32);
    v += __shfl_xor(v,  4, 32);
    v += __shfl_xor(v,  2, 32);
    v += __shfl_xor(v,  1, 32);
    __syncthreads();                         // protect s from previous use
    if ((threadIdx.x & 31) == 0) s[threadIdx.x >> 5] = v;
    __syncthreads();
    return s[0] + s[1] + s[2] + s[3];
}

// ---------------------------------------------------------------------------
// Fused BN (train stats over batch, per timestep) + IndRNN scan + concat-write
//   Y: f32 [8192, O]  (row = t*128 + b)
//   writes bf16 h into X[row*ldx + off + o]
// grid = O blocks, 128 threads (one per batch element)
// ---------------------------------------------------------------------------
__global__ void __launch_bounds__(128)
bn_indrnn_kernel(const float* __restrict__ Y, int O,
                 const float* __restrict__ g, const float* __restrict__ beta,
                 const float* __restrict__ u,
                 __bf16* __restrict__ X, int ldx, int off) {
    __shared__ float s[4];
    const int o = blockIdx.x;
    const int b = threadIdx.x;
    const float gg = g[o], bb = beta[o], uu = u[o];
    float h = 0.0f;
    for (int t = 0; t < 64; ++t) {
        size_t row = (size_t)(t * 128 + b);
        float y   = Y[row * O + o];
        float m   = blockSum128(y, s) * (1.0f / 128.0f);
        float d   = y - m;
        float var = blockSum128(d * d, s) * (1.0f / 128.0f);
        float yn  = d * rsqrtf(var + 1e-5f) * gg + bb;
        h = fmaxf(fmaf(uu, h, yn), 0.0f);
        X[row * ldx + off + o] = (__bf16)h;
    }
}

// ---------------------------------------------------------------------------
// extra_bn: BN only (bf16 in, bf16 out)
// ---------------------------------------------------------------------------
__global__ void __launch_bounds__(128)
bn_only_kernel(const __bf16* __restrict__ Xin, int ldin,
               const float* __restrict__ g, const float* __restrict__ beta,
               __bf16* __restrict__ Xout, int ldout) {
    __shared__ float s[4];
    const int o = blockIdx.x;
    const int b = threadIdx.x;
    const float gg = g[o], bb = beta[o];
    for (int t = 0; t < 64; ++t) {
        size_t row = (size_t)(t * 128 + b);
        float y   = (float)Xin[row * ldin + o];
        float m   = blockSum128(y, s) * (1.0f / 128.0f);
        float d   = y - m;
        float var = blockSum128(d * d, s) * (1.0f / 128.0f);
        Xout[row * ldout + o] = (__bf16)(d * rsqrtf(var + 1e-5f) * gg + bb);
    }
}

// ---------------------------------------------------------------------------
// Host orchestration
// ---------------------------------------------------------------------------
static inline int pad32(int x) { return (x + 31) & ~31; }
static inline int pad64(int x) { return (x + 63) & ~63; }

static void launch_gemm(const __bf16* A, int lda, const __bf16* Bw, int ldb,
                        const float* bias, float* C, int ldc,
                        int M, int N, int K, hipStream_t stream) {
    int waves = (M / 64) * ((N + 63) / 64);
    int blocks = (waves + 3) / 4;                       // 4 waves (128 thr) / block
    gemm_wmma_kernel<<<blocks, 128, 0, stream>>>(A, lda, Bw, ldb, bias, C, ldc, M, N, K);
}

extern "C" void kernel_launch(void* const* d_in, const int* in_sizes, int n_in,
                              void* d_out, int out_size, void* d_ws, size_t ws_size,
                              hipStream_t stream) {
    (void)in_sizes; (void)n_in; (void)out_size; (void)ws_size;
    const int T = 64, B = 128, M = T * B;               // 8192 rows
    const int BLOCKS[4] = {6, 12, 24, 16};
    const int XSTRIDE[4] = {704, 736, 1152, 1088};      // pad32 of block max nf

    // ---- gather inputs -----------------------------------------------------
    const int*   tokens = (const int*)  d_in[0];
    const float* embedt = (const float*)d_in[1];
    struct Unit { const float *W, *b, *g, *beta, *u; int din, dout; size_t wboff; int Kp, Np; };
    Unit units[120];
    {
        int idx = 2, ui = 0, nf = 512;
        auto grab = [&](int din, int dout) {
            Unit& t = units[ui++];
            t.W = (const float*)d_in[idx++]; t.b = (const float*)d_in[idx++];
            t.g = (const float*)d_in[idx++]; t.beta = (const float*)d_in[idx++];
            t.u = (const float*)d_in[idx++];
            t.din = din; t.dout = dout; t.Kp = pad32(din); t.Np = pad64(dout);
        };
        for (int blk = 0; blk < 4; ++blk) {
            for (int j = 0; j < BLOCKS[blk]; ++j) { grab(nf + j * 32, 128); grab(128, 32); }
            nf += BLOCKS[blk] * 32;
            if (blk != 3) { grab(nf, nf / 2); nf /= 2; }
        }
        grab(nf, 512);                                   // extra_fc (nf = 1080)
        const float* extra_g = (const float*)d_in[idx++];
        const float* extra_b = (const float*)d_in[idx++];
        const float* dec_w   = (const float*)d_in[idx++];
        const float* dec_b   = (const float*)d_in[idx++];

        // ---- carve workspace ----------------------------------------------
        char* ws = (char*)d_ws;
        size_t p = 0;
        auto carve = [&](size_t bytes) { void* q = ws + p; p = (p + bytes + 255) & ~(size_t)255; return q; };

        size_t wb_elems = 0;
        for (int i = 0; i < 120; ++i) { units[i].wboff = wb_elems; wb_elems += (size_t)units[i].Np * units[i].Kp; }
        const int decNp = pad64(10000);                  // 10048
        size_t dec_off = wb_elems; wb_elems += (size_t)decNp * 512;
        __bf16* WB = (__bf16*)carve(wb_elems * sizeof(__bf16));
        __bf16* XA = (__bf16*)carve((size_t)M * 1152 * sizeof(__bf16));
        __bf16* XB = (__bf16*)carve((size_t)M * 1152 * sizeof(__bf16));
        __bf16* H  = (__bf16*)carve((size_t)M * 128  * sizeof(__bf16));
        float*  Yf = (float*) carve((size_t)M * 576  * sizeof(float));

        // ---- convert + pad all weights to bf16 -----------------------------
        for (int i = 0; i < 120; ++i) {
            int tot = units[i].Np * units[i].Kp;
            convw_kernel<<<(tot + 255) / 256, 256, 0, stream>>>(units[i].W, WB + units[i].wboff,
                                                                units[i].dout, units[i].din,
                                                                units[i].Np, units[i].Kp);
        }
        {
            int tot = decNp * 512;
            convw_kernel<<<(tot + 255) / 256, 256, 0, stream>>>(dec_w, WB + dec_off, 10000, 512, decNp, 512);
        }

        // ---- embedding -----------------------------------------------------
        embed_kernel<<<M, 256, 0, stream>>>(tokens, embedt, XA, XSTRIDE[0]);

        // ---- dense blocks --------------------------------------------------
        __bf16* Xcur = XA; int ldcur = XSTRIDE[0]; int din = 512; int ui2 = 0;
        for (int blk = 0; blk < 4; ++blk) {
            for (int j = 0; j < BLOCKS[blk]; ++j) {
                Unit& u1 = units[ui2++];                                 // fc1 -> BN -> IndRNN
                launch_gemm(Xcur, ldcur, WB + u1.wboff, u1.Kp, u1.b, Yf, 128, M, 128, din, stream);
                bn_indrnn_kernel<<<128, 128, 0, stream>>>(Yf, 128, u1.g, u1.beta, u1.u, H, 128, 0);
                Unit& u2 = units[ui2++];                                 // fc2 -> BN -> IndRNN
                launch_gemm(H, 128, WB + u2.wboff, u2.Kp, u2.b, Yf, 32, M, 32, 128, stream);
                bn_indrnn_kernel<<<32, 128, 0, stream>>>(Yf, 32, u2.g, u2.beta, u2.u, Xcur, ldcur, din);
                din += 32;                                               // dense concat
            }
            if (blk != 3) {                                              // transition
                Unit& tr = units[ui2++];
                launch_gemm(Xcur, ldcur, WB + tr.wboff, tr.Kp, tr.b, Yf, tr.dout, M, tr.dout, din, stream);
                __bf16* Xnext = (Xcur == XA) ? XB : XA;
                int ldn = XSTRIDE[blk + 1];
                bn_indrnn_kernel<<<tr.dout, 128, 0, stream>>>(Yf, tr.dout, tr.g, tr.beta, tr.u, Xnext, ldn, 0);
                Xcur = Xnext; ldcur = ldn; din = tr.dout;
            }
        }

        // ---- extra_fc unit (1080 -> 512) -----------------------------------
        Unit& ue = units[ui2++];
        launch_gemm(Xcur, ldcur, WB + ue.wboff, ue.Kp, ue.b, Yf, 512, M, 512, din, stream);
        __bf16* Xh = (Xcur == XA) ? XB : XA;
        bn_indrnn_kernel<<<512, 128, 0, stream>>>(Yf, 512, ue.g, ue.beta, ue.u, Xh, 512, 0);

        // ---- extra_bn ------------------------------------------------------
        __bf16* Xfin = (Xh == XA) ? XB : XA;
        bn_only_kernel<<<512, 128, 0, stream>>>(Xh, 512, extra_g, extra_b, Xfin, 512);

        // ---- decoder: [8192,512] x [512,10000]^T + bias -> d_out f32 -------
        launch_gemm(Xfin, 512, WB + dec_off, 512, dec_b, (float*)d_out, 10000, M, 10000, 512, stream);
    }
}
